// PrunableAttention_26079041421991
// MI455X (gfx1250) — matrix operand order repaired and checked by reading, not compiled, not run on hardware
//
#include <hip/hip_runtime.h>

#define HEADS 16
#define DIMH  64
#define BATCH 4
#define SEQ   2048
#define DIM   1024
#define INNER (HEADS*DIMH)   // 1024
#define QKVC  (3*INNER)      // 3072
#define ROWS  (BATCH*SEQ)    // 8192
#define NKB   (SEQ/32)       // 64 key chunks of 32
#define QK_SCALE 0.125f      // 64^-0.5

typedef __attribute__((ext_vector_type(16))) _Float16     v16h;
typedef __attribute__((ext_vector_type(8)))  float        v8f;
typedef __attribute__((ext_vector_type(4)))  unsigned int v4u;
typedef __attribute__((ext_vector_type(4)))  int          v4i;
typedef __attribute__((ext_vector_type(8)))  int          v8i;

union ABu { v16h h; unsigned int u[8]; };

// --- WMMA operand loaders (layouts per CDNA5 ISA 7.12.2, wave32) ---

// A operand: 16x32 f16, memory row-major [M, K], row stride ld.
// lane L: row=L%16, g=L/16; VGPR v holds K pair: v<4: 2v+8g ; v>=4: 16+2(v-4)+8g
static __device__ __forceinline__ v16h load_a16x32(const _Float16* p, int ld) {
  const int lane = threadIdx.x & 31;
  const _Float16* r = p + (lane & 15) * ld + 8 * (lane >> 4);
  ABu u;
  u.u[0] = *(const unsigned int*)(r + 0);
  u.u[1] = *(const unsigned int*)(r + 2);
  u.u[2] = *(const unsigned int*)(r + 4);
  u.u[3] = *(const unsigned int*)(r + 6);
  u.u[4] = *(const unsigned int*)(r + 16);
  u.u[5] = *(const unsigned int*)(r + 18);
  u.u[6] = *(const unsigned int*)(r + 20);
  u.u[7] = *(const unsigned int*)(r + 22);
  return u.h;
}

// B operand: 32x16 (KxN) with memory holding B^T row-major [N, K], stride ld.
static __device__ __forceinline__ v16h load_bT32x16(const _Float16* p, int ld) {
  const int lane = threadIdx.x & 31;
  const _Float16* r = p + (lane & 15) * ld + 8 * (lane >> 4);
  ABu u;
#pragma unroll
  for (int v = 0; v < 8; ++v) u.u[v] = *(const unsigned int*)(r + (v < 4 ? 2 * v : 16 + 2 * (v - 4)));
  return u.h;
}

// B operand: 32x16 (KxN), memory row-major [K, N], stride ld.
static __device__ __forceinline__ v16h load_b32x16(const _Float16* p, int ld) {
  const int lane = threadIdx.x & 31;
  const int n = lane & 15, g = lane >> 4;
  ABu u;
#pragma unroll
  for (int v = 0; v < 8; ++v) {
    const int k = (v < 4 ? 2 * v : 16 + 2 * (v - 4)) + 8 * g;
    u.h[2 * v]     = p[k * ld + n];
    u.h[2 * v + 1] = p[(k + 1) * ld + n];
  }
  return u.h;
}

static __device__ __forceinline__ v8f wmma_f16(v16h a, v16h b, v8f c) {
  return __builtin_amdgcn_wmma_f32_16x16x32_f16(false, a, false, b, (short)0, c,
                                                false, false);
}

// --- TDM: async tensor load of a 32x64 f16 tile (one K/V chunk) to LDS ---
// D# per CDNA5 ISA ch.8: group0 = {count,lds_addr,global_addr,type=2},
// group1 = {data_size=2B, tensor 2048x64 stride 64, tile 32x64}.
static __device__ __forceinline__ void tdm_load_tile32x64(const _Float16* gsrc,
                                                          _Float16* ldst) {
#if __has_builtin(__builtin_amdgcn_tensor_load_to_lds)
  const unsigned long long ga = (unsigned long long)(uintptr_t)gsrc;
  const unsigned la = (unsigned)(uintptr_t)ldst;   // flat LDS addr[31:0] = LDS offset
  v4u g0;
  g0[0] = 1u;                                        // count=1 descriptor
  g0[1] = la;                                        // lds_addr
  g0[2] = (unsigned)(ga & 0xFFFFFFFFull);            // global_addr[31:0]
  g0[3] = (unsigned)((ga >> 32) & 0x1FFFFFFull) | (2u << 30);  // [56:32] | type=2
  v8i g1;
  g1[0] = 0x00010000;            // workgroup_mask=0, data_size=1 (2 bytes)
  g1[1] = (int)(DIMH << 16);     // tensor_dim0 = 64 (low 16 in [31:16])
  g1[2] = (int)(SEQ << 16);      // tensor_dim0 hi=0 | tensor_dim1 = 2048
  g1[3] = (int)(DIMH << 16);     // tensor_dim1 hi=0 | tile_dim0 = 64
  g1[4] = 32;                    // tile_dim1 = 32, tile_dim2 = 0
  g1[5] = DIMH;                  // tensor_dim0_stride = 64
  g1[6] = 0;
  g1[7] = 0;
  v4i gz = {0, 0, 0, 0};
#if defined(__clang_major__) && __clang_major__ >= 23
  v8i z8 = {0, 0, 0, 0, 0, 0, 0, 0};
  __builtin_amdgcn_tensor_load_to_lds(g0, g1, gz, gz, z8, 0);
#else
  __builtin_amdgcn_tensor_load_to_lds(g0, g1, gz, gz, 0);
#endif
#else
  // Fallback: synchronous wave copy (4 KB = 32 lanes x 32 dwords)
  const int lane = threadIdx.x & 31;
#pragma unroll
  for (int i = 0; i < 32; ++i)
    ((unsigned*)ldst)[lane + 32 * i] = ((const unsigned*)gsrc)[lane + 32 * i];
#endif
}

static __device__ __forceinline__ void tdm_wait_2() {
#if __has_builtin(__builtin_amdgcn_tensor_load_to_lds)
  __builtin_amdgcn_s_wait_tensorcnt(2);
#endif
}
static __device__ __forceinline__ void tdm_wait_0() {
#if __has_builtin(__builtin_amdgcn_tensor_load_to_lds)
  __builtin_amdgcn_s_wait_tensorcnt(0);
#endif
}

// --- K0: precision conversion / weight transpose ---

__global__ void cvt_f32_f16(const float* __restrict__ s, _Float16* __restrict__ d, int n) {
  int i = blockIdx.x * blockDim.x + threadIdx.x;
  if (i < n) d[i] = (_Float16)s[i];
}

__global__ void cvt_transpose(const float* __restrict__ s, _Float16* __restrict__ d,
                              int rows, int cols) {
  int i = blockIdx.x * blockDim.x + threadIdx.x;
  if (i < rows * cols) {
    int r = i / cols, c = i - r * cols;
    d[c * rows + r] = (_Float16)s[i];
  }
}

// --- K1: QKV GEMM. One wave = 32x64 tile (2 A-tiles reuse each B-tile) ---

__global__ void __launch_bounds__(256)
qkv_gemm_kernel(const _Float16* __restrict__ Xh,
                const _Float16* __restrict__ WqkvT,
                _Float16* __restrict__ Q,
                _Float16* __restrict__ K,
                _Float16* __restrict__ V) {
  const int wave    = (blockIdx.x * blockDim.x + threadIdx.x) >> 5;
  const int rowTile = wave / (QKVC / 64);     // 0..255 (32 rows each)
  const int colGrp  = wave % (QKVC / 64);     // 0..47  (64 cols each)
  const int lane = threadIdx.x & 31;
  const int g = lane >> 4, col = lane & 15;

  v8f acc[2][4] = {};
  const _Float16* a_base = Xh + (size_t)rowTile * 32 * DIM;
#pragma unroll 1
  for (int k = 0; k < DIM; k += 32) {
    v16h a0 = load_a16x32(a_base + k, DIM);
    v16h a1 = load_a16x32(a_base + 16 * DIM + k, DIM);
#pragma unroll
    for (int t = 0; t < 4; ++t) {
      v16h b = load_bT32x16(WqkvT + (size_t)(colGrp * 64 + t * 16) * DIM + k, DIM);
      acc[0][t] = wmma_f16(a0, b, acc[0][t]);
      acc[1][t] = wmma_f16(a1, b, acc[1][t]);
    }
  }
#pragma unroll
  for (int i = 0; i < 2; ++i)
#pragma unroll
    for (int t = 0; t < 4; ++t)
#pragma unroll
      for (int r = 0; r < 8; ++r) {
        const int row = rowTile * 32 + i * 16 + r + 8 * g;
        const int c   = colGrp * 64 + t * 16 + col;
        const int which = c >> 10, ic = c & 1023;
        const int h = ic >> 6, d = ic & 63;
        const int n = row & (SEQ - 1), bb = row >> 11;
        _Float16* dst = (which == 0) ? Q : ((which == 1) ? K : V);
        dst[(((size_t)(bb * HEADS + h) * SEQ) + n) * DIMH + d] = (_Float16)acc[i][t][r];
      }
}

// --- K2: flash attention. 4 waves x 16 queries per block; K/V chunks staged
//         in LDS via TDM, double-buffered on TENSORcnt ---

__global__ void __launch_bounds__(128)
attn_kernel(const _Float16* __restrict__ Q,
            const _Float16* __restrict__ K,
            const _Float16* __restrict__ V,
            _Float16* __restrict__ O) {
  __shared__ _Float16 Klds[2][32 * DIMH];   // 2 x 4 KB
  __shared__ _Float16 Vlds[2][32 * DIMH];   // 2 x 4 KB
  __shared__ _Float16 Plds[4][16 * 32];     // per-wave P bounce

  const int bh = blockIdx.x >> 5;           // b*16+h (SEQ/64 = 32 qtiles per bh)
  const int qt = blockIdx.x & 31;           // 64-query tile
  const int wid = threadIdx.x >> 5;
  const int lane = threadIdx.x & 31;
  const int g = lane >> 4, col = lane & 15;

  const _Float16* qbase = Q + ((size_t)bh * SEQ + qt * 64 + wid * 16) * DIMH;
  const v16h qa0 = load_a16x32(qbase, DIMH);       // d = 0..31
  const v16h qa1 = load_a16x32(qbase + 32, DIMH);  // d = 32..63

  float m[8], l[8];
#pragma unroll
  for (int r = 0; r < 8; ++r) { m[r] = -3.0e38f; l[r] = 0.0f; }
  v8f o[4] = {};

  const _Float16* kb_base = K + (size_t)bh * SEQ * DIMH;
  const _Float16* vb_base = V + (size_t)bh * SEQ * DIMH;

  if (wid == 0) {                                   // prime buffer 0
    tdm_load_tile32x64(kb_base, Klds[0]);
    tdm_load_tile32x64(vb_base, Vlds[0]);
  }

  for (int kb = 0; kb < NKB; ++kb) {
    const int buf = kb & 1;
    __syncthreads();  // all waves done reading buf^1 (iteration kb-1)
    if (wid == 0) {
      if (kb + 1 < NKB) {
        tdm_load_tile32x64(kb_base + (size_t)(kb + 1) * 32 * DIMH, Klds[buf ^ 1]);
        tdm_load_tile32x64(vb_base + (size_t)(kb + 1) * 32 * DIMH, Vlds[buf ^ 1]);
        tdm_wait_2();   // chunk kb (2 older tensor ops) complete
      } else {
        tdm_wait_0();
      }
    }
    __syncthreads();  // chunk kb visible to all waves

    // S = Q K^T for 32 keys (K chunk rows [n,d] in LDS == B^T layout)
    const _Float16* kp = &Klds[buf][0];
    v8f s0 = {}, s1 = {};
    s0 = wmma_f16(qa0, load_bT32x16(kp, DIMH), s0);
    s0 = wmma_f16(qa1, load_bT32x16(kp + 32, DIMH), s0);
    s1 = wmma_f16(qa0, load_bT32x16(kp + 16 * DIMH, DIMH), s1);
    s1 = wmma_f16(qa1, load_bT32x16(kp + 16 * DIMH + 32, DIMH), s1);

    // online softmax: each row's 16 scores live across 16 lanes of one VGPR
    float resc[8];
#pragma unroll
    for (int r = 0; r < 8; ++r) {
      float a0 = s0[r] * QK_SCALE, a1 = s1[r] * QK_SCALE;
      float mx = fmaxf(a0, a1);
#pragma unroll
      for (int w = 1; w < 16; w <<= 1) mx = fmaxf(mx, __shfl_xor(mx, w, 16));
      const float mn = fmaxf(m[r], mx);
      resc[r] = __expf(m[r] - mn);
      m[r] = mn;
      const float p0 = __expf(a0 - mn);
      const float p1 = __expf(a1 - mn);
      float rs = p0 + p1;
#pragma unroll
      for (int w = 1; w < 16; w <<= 1) rs += __shfl_xor(rs, w, 16);
      l[r] = l[r] * resc[r] + rs;
      s0[r] = p0; s1[r] = p1;
    }
#pragma unroll
    for (int t = 0; t < 4; ++t)
#pragma unroll
      for (int r = 0; r < 8; ++r) o[t][r] *= resc[r];

    // bounce P through per-wave LDS: C layout -> A-operand layout
#pragma unroll
    for (int r = 0; r < 8; ++r) {
      Plds[wid][(r + 8 * g) * 32 + col]      = (_Float16)s0[r];
      Plds[wid][(r + 8 * g) * 32 + 16 + col] = (_Float16)s1[r];
    }
    asm volatile("" ::: "memory");  // wave-internal LDS ordering (LDS is in-order)
    v16h pa;
    {
      const _Float16* pr = &Plds[wid][(lane & 15) * 32 + 8 * g];
      ABu u;
      u.u[0] = *(const unsigned int*)(pr + 0);
      u.u[1] = *(const unsigned int*)(pr + 2);
      u.u[2] = *(const unsigned int*)(pr + 4);
      u.u[3] = *(const unsigned int*)(pr + 6);
      u.u[4] = *(const unsigned int*)(pr + 16);
      u.u[5] = *(const unsigned int*)(pr + 18);
      u.u[6] = *(const unsigned int*)(pr + 20);
      u.u[7] = *(const unsigned int*)(pr + 22);
      pa = u.h;
    }
    // O += P V : V chunk [keys, d] in LDS -> plain B layout
    const _Float16* vp = &Vlds[buf][0];
#pragma unroll
    for (int t = 0; t < 4; ++t)
      o[t] = wmma_f16(pa, load_b32x16(vp + t * 16, DIMH), o[t]);
  }

  // normalize and store to [bn, inner] f16 (undoes head transpose)
  const int bb = bh >> 4, h = bh & 15;
#pragma unroll
  for (int r = 0; r < 8; ++r) {
    const float inv = 1.0f / l[r];
    const int n = qt * 64 + wid * 16 + r + 8 * g;
#pragma unroll
    for (int t = 0; t < 4; ++t)
      O[((size_t)bb * SEQ + n) * INNER + h * DIMH + t * 16 + col] =
          (_Float16)(o[t][r] * inv);
  }
}

// --- K3: output projection + bias, f32 out. One wave = 32x64 tile ---

__global__ void __launch_bounds__(256)
proj_gemm_kernel(const _Float16* __restrict__ Oh,
                 const _Float16* __restrict__ WprojT,
                 const float* __restrict__ bias,
                 float* __restrict__ out) {
  const int wave    = (blockIdx.x * blockDim.x + threadIdx.x) >> 5;
  const int rowTile = wave / (DIM / 64);   // 0..255
  const int colGrp  = wave % (DIM / 64);   // 0..15
  const int lane = threadIdx.x & 31;
  const int g = lane >> 4, col = lane & 15;

  v8f acc[2][4] = {};
  const _Float16* a_base = Oh + (size_t)rowTile * 32 * INNER;
#pragma unroll 1
  for (int k = 0; k < INNER; k += 32) {
    v16h a0 = load_a16x32(a_base + k, INNER);
    v16h a1 = load_a16x32(a_base + 16 * INNER + k, INNER);
#pragma unroll
    for (int t = 0; t < 4; ++t) {
      v16h b = load_bT32x16(WprojT + (size_t)(colGrp * 64 + t * 16) * INNER + k, INNER);
      acc[0][t] = wmma_f16(a0, b, acc[0][t]);
      acc[1][t] = wmma_f16(a1, b, acc[1][t]);
    }
  }
#pragma unroll
  for (int t = 0; t < 4; ++t) {
    const int c = colGrp * 64 + t * 16 + col;
    const float bv = bias[c];
#pragma unroll
    for (int i = 0; i < 2; ++i)
#pragma unroll
      for (int r = 0; r < 8; ++r) {
        const int row = rowTile * 32 + i * 16 + r + 8 * g;
        out[(size_t)row * DIM + c] = acc[i][t][r] + bv;
      }
  }
}

extern "C" void kernel_launch(void* const* d_in, const int* in_sizes, int n_in,
                              void* d_out, int out_size, void* d_ws, size_t ws_size,
                              hipStream_t stream) {
  const float* X     = (const float*)d_in[0];  // [4,2048,1024]
  const float* Wqkv  = (const float*)d_in[1];  // [1024,3072]
  const float* Wproj = (const float*)d_in[2];  // [1024,1024]
  const float* bproj = (const float*)d_in[3];  // [1024]
  float* out = (float*)d_out;

  char* ws = (char*)d_ws;
  const size_t MB = 1024 * 1024;
  _Float16* Xh     = (_Float16*)(ws + 0 * MB);   // 16 MB
  _Float16* WqkvT  = (_Float16*)(ws + 16 * MB);  //  6 MB
  _Float16* WprojT = (_Float16*)(ws + 22 * MB);  //  2 MB
  _Float16* Qw     = (_Float16*)(ws + 24 * MB);  // 16 MB
  _Float16* Kw     = (_Float16*)(ws + 40 * MB);  // 16 MB
  _Float16* Vw     = (_Float16*)(ws + 56 * MB);  // 16 MB
  _Float16* Ow     = (_Float16*)(ws + 72 * MB);  // 16 MB

  const int nX = ROWS * DIM;
  cvt_f32_f16<<<(nX + 255) / 256, 256, 0, stream>>>(X, Xh, nX);
  cvt_transpose<<<(DIM * QKVC + 255) / 256, 256, 0, stream>>>(Wqkv, WqkvT, DIM, QKVC);
  cvt_transpose<<<(INNER * DIM + 255) / 256, 256, 0, stream>>>(Wproj, WprojT, INNER, DIM);

  {
    const int waves = (ROWS / 32) * (QKVC / 64);  // 12288
    qkv_gemm_kernel<<<waves / 8, 256, 0, stream>>>(Xh, WqkvT, Qw, Kw, Vw);
  }
  attn_kernel<<<BATCH * HEADS * (SEQ / 64), 128, 0, stream>>>(Qw, Kw, Vw, Ow);
  {
    const int waves = (ROWS / 32) * (DIM / 64);   // 4096
    proj_gemm_kernel<<<waves / 8, 256, 0, stream>>>(Ow, WprojT, bproj, out);
  }
}
